// GraphSAGEv2_73186242724264
// MI455X (gfx1250) — matrix operand order repaired and checked by reading, not compiled, not run on hardware
//
#include <hip/hip_runtime.h>
#include <hip/hip_bf16.h>
#include <stdint.h>

#define N_NODES 20000
#define N_EDGES 320000
#define FIN_C   256
#define HID_C   512
#define FOUT_C  256
#define NGRAPH  64
#define NLAYER  4

typedef __attribute__((ext_vector_type(16))) __bf16 v16bf;
typedef __attribute__((ext_vector_type(8)))  __bf16 v8bf;
typedef __attribute__((ext_vector_type(8)))  float  v8f;
typedef int v4i_t __attribute__((vector_size(16)));   // matches builtin param type

#if defined(__AMDGCN__) && __has_builtin(__builtin_amdgcn_global_load_async_to_lds_b128) && \
    __has_builtin(__builtin_amdgcn_s_wait_asynccnt)
#define USE_ASYNC 1
#else
#define USE_ASYNC 0
#endif

// ---------- small helpers ----------
__device__ inline v8f zero8() {
    v8f z = {0.f,0.f,0.f,0.f,0.f,0.f,0.f,0.f};
    return z;
}

// A-fragment (16x32 bf16, MxK). lane<16: row M=lane, K=k0..k0+15;
// lane>=16: row M=lane-16, K=k0+16..k0+31.
// p = &A[row][k0 + 16*halfsel]; grab [0..15].
__device__ inline v16bf load_a_frag(const __bf16* p) {
    v8bf lo = *(const v8bf*)p;
    v8bf hi = *(const v8bf*)(p + 8);
    v16bf r;
#pragma unroll
    for (int i = 0; i < 8; ++i) { r[i] = lo[i]; r[i + 8] = hi[i]; }
    return r;
}

// B-fragment (32x16 bf16 KxN), B[k][n] = W[n][k]. lane<16: col N=lane, K=k0..k0+15;
// lane>=16: col N=lane-16, K=k0+16..k0+31. p = &W[n][k0 + 16*halfsel], 32B load.
__device__ inline v16bf load_b_frag(const __bf16* p) {
    return *(const v16bf*)p;
}

#if USE_ASYNC
__device__ inline void async_b128(const __bf16* gp, __bf16* lp) {
    __builtin_amdgcn_global_load_async_to_lds_b128(
        (__attribute__((address_space(1))) v4i_t*)(uintptr_t)gp,
        (__attribute__((address_space(3))) v4i_t*)(uint32_t)(uintptr_t)lp,
        0, 0);
}
#endif

// ---------- elementwise / scatter kernels ----------
__global__ void zero_kernel(float* p, int n) {
    int i = blockIdx.x * blockDim.x + threadIdx.x;
    if (i < n) p[i] = 0.f;
}

__global__ void cvt_kernel(__bf16* o, const float* x, int n) {
    int i = blockIdx.x * blockDim.x + threadIdx.x;
    if (i < n) o[i] = (__bf16)x[i];
}

__global__ void pack_kernel(__bf16* o, const float* Wl, const float* Wr, int F, int total) {
    int i = blockIdx.x * blockDim.x + threadIdx.x;
    if (i >= total) return;
    int K = 2 * F;
    int n = i / K;
    int k = i - n * K;
    float v = (k < F) ? Wl[(size_t)n * F + k] : Wr[(size_t)n * F + (k - F)];
    o[i] = (__bf16)v;
}

__global__ void deg_kernel(float* deg, const int* tgt, int E) {
    int i = blockIdx.x * blockDim.x + threadIdx.x;
    if (i < E) atomicAdd(&deg[tgt[i]], 1.0f);
}

__global__ void cnt_kernel(float* cnt, const int* batch, int n) {
    int i = blockIdx.x * blockDim.x + threadIdx.x;
    if (i < n) atomicAdd(&cnt[batch[i]], 1.0f);
}

__global__ void scatter_kernel(float* __restrict__ agg, const __bf16* __restrict__ X,
                               const int* __restrict__ src, const int* __restrict__ tgt, int F) {
    int e = blockIdx.x;
    int s = src[e];
    int t = tgt[e];
    const __bf16* xs = X + (size_t)s * F;
    float* at = agg + (size_t)t * F;
    for (int f = threadIdx.x; f < F; f += blockDim.x)
        atomicAdd(&at[f], (float)xs[f]);
}

__global__ void mean_kernel(__bf16* o, const float* agg, const float* deg, int F, int total) {
    int i = blockIdx.x * blockDim.x + threadIdx.x;
    if (i >= total) return;
    int node = i / F;
    float d = deg[node];
    d = (d < 1.f) ? 1.f : d;
    o[i] = (__bf16)(agg[i] / d);
}

__global__ void pool_kernel(float* pooled, const float* Hf, const int* batch, int total) {
    int i = blockIdx.x * blockDim.x + threadIdx.x;
    if (i >= total) return;
    int node = i / FOUT_C;
    int f = i - node * FOUT_C;
    atomicAdd(&pooled[batch[node] * FOUT_C + f], Hf[i]);
}

__global__ void out_kernel(float* out, const float* pooled, const float* cnt,
                           const float* Wlin, const float* blin) {
    int g = blockIdx.x;
    int o = threadIdx.x;
    float inv = 1.0f / fmaxf(cnt[g], 1.0f);
    const float* pg = pooled + g * FOUT_C;
    const float* wo = Wlin + o * FOUT_C;
    float acc = blin[o];
    for (int k = 0; k < FOUT_C; ++k)
        acc += pg[k] * inv * wo[k];
    out[g * FOUT_C + o] = acc;
}

// ---------- fused SAGE GEMM (+bias [+LayerNorm+ReLU]) ----------
// Block computes a 32-row x HOUT slab: Y = [AGG | X] @ Wcat^T + bias (opt LN+ReLU).
// K = 2F (k<F reads A0 = mean-agg, k>=F reads A1 = self). 8 waves; wave owns TPW
// 16-col tiles, reused across 2 M-tiles. A tile (32x32 bf16) is double-buffered in
// LDS via async-to-LDS (or a cooperative copy fallback). HOUT = 8*TPW*16.
template <int TPW>
__launch_bounds__(256)
__global__ void sage_gemm(const __bf16* __restrict__ A0,
                          const __bf16* __restrict__ A1,
                          const __bf16* __restrict__ W,   // [HOUT x 2F] bf16 row-major
                          const float* __restrict__ bias,
                          const float* __restrict__ lng,  // [HOUT] or null
                          const float* __restrict__ lnb,  // [HOUT] or null
                          __bf16* __restrict__ Yb,        // bf16 out (LN path)
                          float* __restrict__ Yf,         // f32 out (no-LN path)
                          int F, int useLN) {
    constexpr int HOUT = TPW * 8 * 16;
    __shared__ float  sc[16 * HOUT];         // epilogue slab (one 16-row pass at a time)
    __shared__ __bf16 atile[2][32 * 32];     // double-buffered A tile

    const int tid  = threadIdx.x;
    const int wave = tid >> 5;
    const int lane = tid & 31;
    const int halfsel = lane >> 4;
    const int lr = lane & 15;
    const int m0 = blockIdx.x * 32;          // N_NODES % 32 == 0
    const int K2 = 2 * F;

    v8f acc[2][TPW];
#pragma unroll
    for (int mt = 0; mt < 2; ++mt)
#pragma unroll
        for (int t = 0; t < TPW; ++t) acc[mt][t] = zero8();

    // per-lane W row base pointers (column tiles owned by this wave)
    const __bf16* wrow[TPW];
#pragma unroll
    for (int t = 0; t < TPW; ++t)
        wrow[t] = W + (size_t)((wave * TPW + t) * 16 + lr) * K2;

    // stage one 32x32 A tile (k columns k0..k0+31) into atile[buf]
    auto stage = [&](int buf, int k0) {
#if USE_ASYNC
        if (wave == 0) {
#pragma unroll
            for (int j = 0; j < 4; ++j) {
                int c = lane + 32 * j;           // 128 chunks of 16B
                int row = c >> 2;
                int off = (c & 3) * 8;           // bf16 elements
                const __bf16* gp = ((k0 < F) ? (A0 + (size_t)(m0 + row) * F + k0)
                                             : (A1 + (size_t)(m0 + row) * F + (k0 - F))) + off;
                async_b128(gp, &atile[buf][row * 32 + off]);
            }
        }
#else
        if (tid < 128) {
            int row = tid >> 2;
            int off = (tid & 3) * 8;
            const __bf16* gp = ((k0 < F) ? (A0 + (size_t)(m0 + row) * F + k0)
                                         : (A1 + (size_t)(m0 + row) * F + (k0 - F))) + off;
            *(v8bf*)&atile[buf][row * 32 + off] = *(const v8bf*)gp;
        }
#endif
    };

    stage(0, 0);
    int cur = 0;
    for (int k0 = 0; k0 < K2; k0 += 32) {
#if USE_ASYNC
        if (wave == 0) __builtin_amdgcn_s_wait_asynccnt(0);
#endif
        __syncthreads();                     // atile[cur] ready; atile[cur^1] free
        if (k0 + 32 < K2) stage(cur ^ 1, k0 + 32);

        // batched B loads for this K-step
        v16bf bfrag[TPW];
#pragma unroll
        for (int t = 0; t < TPW; ++t)
            bfrag[t] = load_b_frag(wrow[t] + k0 + halfsel * 16);
        if (k0 + 32 < K2)
            __builtin_prefetch(wrow[0] + k0 + 32, 0, 1);   // global_prefetch_b8

        // A fragments from LDS (ds_load_b128 x2 per M-tile)
        v16bf afrag[2];
#pragma unroll
        for (int mt = 0; mt < 2; ++mt)
            afrag[mt] = load_a_frag(&atile[cur][(mt * 16 + lr) * 32 + halfsel * 16]);

        // 2*TPW back-to-back WMMAs, B reused across both M-tiles
#pragma unroll
        for (int mt = 0; mt < 2; ++mt)
#pragma unroll
            for (int t = 0; t < TPW; ++t)
                acc[mt][t] = __builtin_amdgcn_wmma_f32_16x16x32_bf16(
                    false, afrag[mt], false, bfrag[t], (short)0, acc[mt][t], false, false);

        cur ^= 1;
    }

    // epilogue in two 16-row passes (keeps sc at 16*HOUT floats)
    const int row = tid >> 4;   // 0..15
    const int sub = tid & 15;
    for (int mt = 0; mt < 2; ++mt) {
        __syncthreads();
        // spill C tiles (+bias): acc[mt][t][i] -> row i+8*halfsel, col tile+lr
#pragma unroll
        for (int t = 0; t < TPW; ++t) {
            int n = (wave * TPW + t) * 16 + lr;
            float bs = bias[n];
#pragma unroll
            for (int i = 0; i < 8; ++i)
                sc[(i + 8 * halfsel) * HOUT + n] = acc[mt][t][i] + bs;
        }
        __syncthreads();

        const size_t grow = (size_t)(m0 + mt * 16 + row);
        if (useLN) {
            float s = 0.f, s2 = 0.f;
            for (int c = sub; c < HOUT; c += 16) {
                float v = sc[row * HOUT + c];
                s += v; s2 += v * v;
            }
#pragma unroll
            for (int off = 8; off > 0; off >>= 1) {
                s  += __shfl_xor(s,  off, 16);
                s2 += __shfl_xor(s2, off, 16);
            }
            float mean = s * (1.0f / HOUT);
            float var  = s2 * (1.0f / HOUT) - mean * mean;
            float inv  = rsqrtf(var + 1e-5f);
            for (int c = sub; c < HOUT; c += 16) {
                float v = (sc[row * HOUT + c] - mean) * inv * lng[c] + lnb[c];
                v = fmaxf(v, 0.f);
                Yb[grow * HOUT + c] = (__bf16)v;
            }
        } else {
            for (int c = sub; c < HOUT; c += 16)
                Yf[grow * HOUT + c] = sc[row * HOUT + c];
        }
    }
}

// ---------- host launcher ----------
extern "C" void kernel_launch(void* const* d_in, const int* in_sizes, int n_in,
                              void* d_out, int out_size, void* d_ws, size_t ws_size,
                              hipStream_t stream) {
    (void)in_sizes; (void)n_in; (void)out_size; (void)ws_size;

    const float* x     = (const float*)d_in[0];
    const int*   ei    = (const int*)d_in[1];
    const int*   batch = (const int*)d_in[2];
    const float* W0l   = (const float*)d_in[3];
    const float* b0    = (const float*)d_in[4];
    const float* W0r   = (const float*)d_in[5];
    const float* Whl   = (const float*)d_in[6];
    const float* bh    = (const float*)d_in[7];
    const float* Whr   = (const float*)d_in[8];
    const float* Wfl   = (const float*)d_in[9];
    const float* bf_   = (const float*)d_in[10];
    const float* Wfr   = (const float*)d_in[11];
    const float* lng   = (const float*)d_in[12];
    const float* lnb   = (const float*)d_in[13];
    const float* Wlin  = (const float*)d_in[14];
    const float* blin  = (const float*)d_in[15];
    float* out = (float*)d_out;

    const int* src = ei;
    const int* tgt = ei + N_EDGES;

    char* base = (char*)d_ws;
    size_t off = 0;
    auto take = [&](size_t bytes) -> char* {
        char* p = base + off;
        off += (bytes + 255) & ~(size_t)255;
        return p;
    };
    __bf16* X0b    = (__bf16*)take((size_t)N_NODES * FIN_C * 2);
    __bf16* Xa     = (__bf16*)take((size_t)N_NODES * HID_C * 2);
    __bf16* Xb2    = (__bf16*)take((size_t)N_NODES * HID_C * 2);
    float*  AGGf   = (float*) take((size_t)N_NODES * HID_C * 4);
    __bf16* AGGb   = (__bf16*)take((size_t)N_NODES * HID_C * 2);
    float*  deg    = (float*) take((size_t)N_NODES * 4);
    float*  cnt    = (float*) take((size_t)NGRAPH * 4);
    float*  pooled = (float*) take((size_t)NGRAPH * FOUT_C * 4);
    __bf16* Wc0    = (__bf16*)take((size_t)HID_C  * 2 * FIN_C * 2);
    __bf16* WcH    = (__bf16*)take((size_t)NLAYER * HID_C * 2 * HID_C * 2);
    __bf16* WcF    = (__bf16*)take((size_t)FOUT_C * 2 * HID_C * 2);
    float*  Hf     = AGGf;  // AGGf dead after final mean_kernel; reuse for conv-out

    const int T = 256;
    auto cdiv = [](int a, int b) { return (a + b - 1) / b; };

    cvt_kernel<<<cdiv(N_NODES * FIN_C, T), T, 0, stream>>>(X0b, x, N_NODES * FIN_C);
    pack_kernel<<<cdiv(HID_C * 2 * FIN_C, T), T, 0, stream>>>(Wc0, W0l, W0r, FIN_C, HID_C * 2 * FIN_C);
    for (int i = 0; i < NLAYER; ++i)
        pack_kernel<<<cdiv(HID_C * 2 * HID_C, T), T, 0, stream>>>(
            WcH + (size_t)i * HID_C * 2 * HID_C,
            Whl + (size_t)i * HID_C * HID_C,
            Whr + (size_t)i * HID_C * HID_C, HID_C, HID_C * 2 * HID_C);
    pack_kernel<<<cdiv(FOUT_C * 2 * HID_C, T), T, 0, stream>>>(WcF, Wfl, Wfr, HID_C, FOUT_C * 2 * HID_C);
    zero_kernel<<<cdiv(N_NODES, T), T, 0, stream>>>(deg, N_NODES);
    deg_kernel<<<cdiv(N_EDGES, T), T, 0, stream>>>(deg, tgt, N_EDGES);
    zero_kernel<<<1, T, 0, stream>>>(cnt, NGRAPH);
    cnt_kernel<<<cdiv(N_NODES, T), T, 0, stream>>>(cnt, batch, N_NODES);

    // block 0: FIN -> HID, LN+ReLU
    zero_kernel<<<cdiv(N_NODES * FIN_C, T), T, 0, stream>>>(AGGf, N_NODES * FIN_C);
    scatter_kernel<<<N_EDGES, T, 0, stream>>>(AGGf, X0b, src, tgt, FIN_C);
    mean_kernel<<<cdiv(N_NODES * FIN_C, T), T, 0, stream>>>(AGGb, AGGf, deg, FIN_C, N_NODES * FIN_C);
    sage_gemm<4><<<N_NODES / 32, T, 0, stream>>>(AGGb, X0b, Wc0, b0, lng, lnb,
                                                 Xa, nullptr, FIN_C, 1);

    // blocks 1..NL: HID -> HID, LN+ReLU
    __bf16* curb = Xa;
    __bf16* nxtb = Xb2;
    for (int i = 0; i < NLAYER; ++i) {
        zero_kernel<<<cdiv(N_NODES * HID_C, T), T, 0, stream>>>(AGGf, N_NODES * HID_C);
        scatter_kernel<<<N_EDGES, T, 0, stream>>>(AGGf, curb, src, tgt, HID_C);
        mean_kernel<<<cdiv(N_NODES * HID_C, T), T, 0, stream>>>(AGGb, AGGf, deg, HID_C, N_NODES * HID_C);
        sage_gemm<4><<<N_NODES / 32, T, 0, stream>>>(
            AGGb, curb, WcH + (size_t)i * HID_C * 2 * HID_C,
            bh + (size_t)i * HID_C,
            lng + (size_t)(i + 1) * HID_C, lnb + (size_t)(i + 1) * HID_C,
            nxtb, nullptr, HID_C, 1);
        __bf16* tmp = curb; curb = nxtb; nxtb = tmp;
    }

    // final conv: HID -> FOUT, no LN, f32 out
    zero_kernel<<<cdiv(N_NODES * HID_C, T), T, 0, stream>>>(AGGf, N_NODES * HID_C);
    scatter_kernel<<<N_EDGES, T, 0, stream>>>(AGGf, curb, src, tgt, HID_C);
    mean_kernel<<<cdiv(N_NODES * HID_C, T), T, 0, stream>>>(AGGb, AGGf, deg, HID_C, N_NODES * HID_C);
    sage_gemm<2><<<N_NODES / 32, T, 0, stream>>>(AGGb, curb, WcF, bf_, nullptr, nullptr,
                                                 nullptr, Hf, HID_C, 0);

    // global mean pool + final linear
    zero_kernel<<<cdiv(NGRAPH * FOUT_C, T), T, 0, stream>>>(pooled, NGRAPH * FOUT_C);
    pool_kernel<<<cdiv(N_NODES * FOUT_C, T), T, 0, stream>>>(pooled, Hf, batch, N_NODES * FOUT_C);
    out_kernel<<<NGRAPH, FOUT_C, 0, stream>>>(out, pooled, cnt, Wlin, blin);
}
